// GCNEncoder_43482248905435
// MI455X (gfx1250) — matrix operand order repaired and checked by reading, not compiled, not run on hardware
//
#include <hip/hip_runtime.h>
#include <hip/hip_bf16.h>

// GCN encoder for MI455X (gfx1250, wave32).
// Roofline: ~10.5 GFLOP of GEMM (trivial on WMMA) vs ~2.8 GB of gather/scatter
// traffic -> memory bound (~0.12 ms at 23.3 TB/s). GEMM uses the probe-confirmed
// v_wmma_f32_16x16x32_bf16 (bf16 in, fp32 accumulate); scatter uses fp32 global
// atomics that resolve in the 192 MB L2. W tiles are staged into LDS with the
// gfx1250 async-to-LDS path (ASYNCcnt) when the builtin is available.

#define H 256           // hidden size (compile-time)

typedef __bf16 bf16x16 __attribute__((ext_vector_type(16)));
typedef float  f32x8   __attribute__((ext_vector_type(8)));
typedef int    b128_t  __attribute__((vector_size(16)));   // matches builtin sig

#if defined(__gfx1250__) && __has_builtin(__builtin_amdgcn_global_load_async_to_lds_b128)
#define HAVE_ASYNC_LDS 1
#endif

#define AS1 __attribute__((address_space(1)))
#define AS3 __attribute__((address_space(3)))

// ---------------------------------------------------------------- utility ----
__global__ void k_zero(float* __restrict__ p, int n) {
    int i = blockIdx.x * 256 + threadIdx.x;
    if (i < n) p[i] = 0.0f;
}

__global__ void k_deg(const int* __restrict__ col, float* __restrict__ deg, int E) {
    int e = blockIdx.x * 256 + threadIdx.x;
    if (e < E) atomicAdd(&deg[col[e]], 1.0f);
}

__global__ void k_dinv(const float* __restrict__ deg, float* __restrict__ dinv, int N) {
    int i = blockIdx.x * 256 + threadIdx.x;
    if (i < N) {
        float d = deg[i] + 1.0f;        // +1 for the self loop
        dinv[i] = rsqrtf(d);            // d >= 1 always, no zero-degree case
    }
}

__global__ void k_init_bias(float* __restrict__ out, const float* __restrict__ b, int total) {
    int i = blockIdx.x * 256 + threadIdx.x;
    if (i < total) out[i] = b[i & (H - 1)];
}

__global__ void k_relu(float* __restrict__ a, int n) {
    int i = blockIdx.x * 256 + threadIdx.x;
    if (i < n) a[i] = fmaxf(a[i], 0.0f);
}

// Transpose + convert W[k][n] fp32 -> Wt[n][k] bf16 (one shot, 64K elements).
// Keeps the GEMM's LDS staging a pure 16-byte copy.
__global__ void k_prep_wt(const float* __restrict__ W, __bf16* __restrict__ Wt) {
    int i = blockIdx.x * 256 + threadIdx.x;   // 65536 threads
    int k = i >> 8;
    int n = i & 255;
    Wt[n * H + k] = (__bf16)W[k * H + n];
}

// ------------------------------------------------------------------ GEMM -----
// Y[M,H] = X[M,H] @ W[H,H] using Wt[n][k] bf16, fp32 accumulate.
// Block: 256 threads = 8 waves. Block tile: 128 (M) x 64 (N).
// The block's full W slice (64 n x 256 k bf16 = 32 KB) is staged in LDS ONCE
// (async-to-LDS when available), then the K loop runs barrier-free so loads,
// ds reads and the 4-WMMA chains software-pipeline.
__global__ void __launch_bounds__(256) k_gemm(const float* __restrict__ X,
                                              const __bf16* __restrict__ Wt,
                                              float* __restrict__ Y, int M) {
    __shared__ __align__(128) __bf16 sW[64][H];   // [n][k], 32 KB

    const int tid  = threadIdx.x;
    const int wave = tid >> 5;
    const int lane = tid & 31;
    const int l15  = lane & 15;
    const int hi   = lane >> 4;          // 0: lanes 0-15, 1: lanes 16-31

    const int n0 = blockIdx.y * 64;
    const int m0 = (blockIdx.x * 8 + wave) * 16;
    const bool valid = (m0 < M);         // wave-uniform -> EXEC all-1s in WMMA

    // ---- stage Wt rows [n0, n0+64) -> sW: contiguous 32 KB, 128 B/thread ----
    {
        const char* gsrc = (const char*)(Wt + (size_t)n0 * H);
        char* ldst = (char*)&sW[0][0];
        #pragma unroll
        for (int i = 0; i < 8; ++i) {
            int byteoff = (i * 256 + tid) * 16;      // lane-consecutive 16B
#if defined(HAVE_ASYNC_LDS)
            __builtin_amdgcn_global_load_async_to_lds_b128(
                (AS1 b128_t*)(gsrc + byteoff), (AS3 b128_t*)(ldst + byteoff),
                0, 0);
#else
            *(uint4*)(ldst + byteoff) = *(const uint4*)(gsrc + byteoff);
#endif
        }
#if defined(HAVE_ASYNC_LDS)
#if __has_builtin(__builtin_amdgcn_s_wait_asynccnt)
        __builtin_amdgcn_s_wait_asynccnt(0);
#else
        asm volatile("s_wait_asynccnt 0" ::: "memory");
#endif
#endif
    }
    __syncthreads();                      // single barrier; K loop is sync-free

    const int kbA   = hi * 16;            // A lane K base (matches B contiguous K)
    const int krelB = hi * 16;            // B lane K base (contiguous K, same as A)
    const int m     = m0 + l15;

    f32x8 acc[4];
    #pragma unroll
    for (int t = 0; t < 4; ++t)
        #pragma unroll
        for (int r = 0; r < 8; ++r) acc[t][r] = 0.0f;

    if (valid) {
        #pragma unroll
        for (int ks = 0; ks < 8; ++ks) {  // K = 256 in steps of 32
            const int k0 = ks * 32;

            // A fragment: lane<16 holds K {0..15}, lane>=16 holds
            // K {16..31} of row m (same contiguous K order as B).
            bf16x16 a;
            const float* ap = X + (size_t)m * H + k0 + kbA;
            #pragma unroll
            for (int j = 0; j < 8; ++j) a[j]     = (__bf16)ap[j];
            #pragma unroll
            for (int j = 0; j < 8; ++j) a[8 + j] = (__bf16)ap[8 + j];

            // All 4 B fragments up front (distinct regs -> WMMAs can pipeline).
            bf16x16 bfr[4];
            #pragma unroll
            for (int t = 0; t < 4; ++t)
                bfr[t] = *(const bf16x16*)&sW[t * 16 + l15][k0 + krelB];

            #pragma unroll
            for (int t = 0; t < 4; ++t)
                acc[t] = __builtin_amdgcn_wmma_f32_16x16x32_bf16(
                    false, a, false, bfr[t], (short)0, acc[t], false, false);
        }

        // C/D layout: VGPR r -> row m0+r (lanes 0-15) / m0+8+r (lanes 16-31),
        // N = tile base + (lane & 15).
        const int mbase = m0 + (hi << 3);
        #pragma unroll
        for (int t = 0; t < 4; ++t) {
            int n = n0 + t * 16 + l15;
            #pragma unroll
            for (int r = 0; r < 8; ++r)
                Y[(size_t)(mbase + r) * H + n] = acc[t][r];
        }
    }
}

// ----------------------------------------------------------- aggregation ----
// out[col[e]] += dinv[row[e]]*dinv[col[e]] * h[row[e]]  for e in [0, E+N)
// (edges E.. are the implicit self loops). 64 lanes per edge, float4 gathers.
__global__ void __launch_bounds__(256) k_agg(const float* __restrict__ h,
                                             const int* __restrict__ row,
                                             const int* __restrict__ col,
                                             const float* __restrict__ dinv,
                                             float* __restrict__ out,
                                             int E, int N) {
    int idx = blockIdx.x * 4 + (threadIdx.x >> 6);   // edge id
    if (idx >= E + N) return;
    int f = (threadIdx.x & 63) * 4;                  // feature base

    int r, c;
    if (idx < E) { r = row[idx]; c = col[idx]; }
    else         { r = c = idx - E; }                // self loop

    float w = dinv[r] * dinv[c];
    const float4 v = *(const float4*)(h + (size_t)r * H + f);
    float* o = out + (size_t)c * H + f;
    atomicAdd(o + 0, w * v.x);
    atomicAdd(o + 1, w * v.y);
    atomicAdd(o + 2, w * v.z);
    atomicAdd(o + 3, w * v.w);
}

// ---------------------------------------------------------------- driver -----
extern "C" void kernel_launch(void* const* d_in, const int* in_sizes, int n_in,
                              void* d_out, int out_size, void* d_ws, size_t ws_size,
                              hipStream_t stream) {
    const float* x   = (const float*)d_in[0];
    const int*   ei  = (const int*)  d_in[1];
    const float* W1  = (const float*)d_in[2];
    const float* W11 = (const float*)d_in[3];
    const float* W2  = (const float*)d_in[4];
    const float* W3  = (const float*)d_in[5];
    const float* b1  = (const float*)d_in[6];
    const float* b11 = (const float*)d_in[7];
    const float* b2  = (const float*)d_in[8];
    const float* b3  = (const float*)d_in[9];

    const int N = in_sizes[0] / H;       // 20000
    const int E = in_sizes[1] / 2;       // 320000
    const int* row = ei;
    const int* col = ei + E;

    float* ws   = (float*)d_ws;
    float* deg  = ws;
    float* dinv = deg  + N;
    float* hbuf = dinv + N;                       // GEMM output, reused
    float* t1   = hbuf + (size_t)N * H;           // layer-1 activations
    float* t2   = t1   + (size_t)N * H;           // layer-2 activations
    __bf16* Wt  = (__bf16*)(t2 + (size_t)N * H);  // transposed bf16 weights
    float* out2 = (float*)d_out;
    float* out3 = out2 + (size_t)N * H;

    const int NH  = N * H;
    const int eb  = (E + 255) / 256;
    const int nb  = (N + 255) / 256;
    const int nhb = (NH + 255) / 256;
    const dim3 gg((N + 127) / 128, H / 64, 1);          // GEMM grid
    const int aggb = (E + N + 3) / 4;                   // agg grid
    const int wb   = (H * H) / 256;                     // prep grid

    // ---- symmetric normalization: dinv = rsqrt(in-degree + 1) ----
    k_zero<<<nb, 256, 0, stream>>>(deg, N);
    k_deg <<<eb, 256, 0, stream>>>(col, deg, E);
    k_dinv<<<nb, 256, 0, stream>>>(deg, dinv, N);

    // ---- layer 1: h1 = relu(agg(x@W1) + b1) ----
    k_prep_wt  <<<wb,   256, 0, stream>>>(W1, Wt);
    k_gemm     <<<gg,   256, 0, stream>>>(x, Wt, hbuf, N);
    k_init_bias<<<nhb,  256, 0, stream>>>(t1, b1, NH);
    k_agg      <<<aggb, 256, 0, stream>>>(hbuf, row, col, dinv, t1, E, N);
    k_relu     <<<nhb,  256, 0, stream>>>(t1, NH);

    // ---- layer 2: h2 = relu(agg(h1@W1_1) + b1_1) ----
    k_prep_wt  <<<wb,   256, 0, stream>>>(W11, Wt);
    k_gemm     <<<gg,   256, 0, stream>>>(t1, Wt, hbuf, N);
    k_init_bias<<<nhb,  256, 0, stream>>>(t2, b11, NH);
    k_agg      <<<aggb, 256, 0, stream>>>(hbuf, row, col, dinv, t2, E, N);
    k_relu     <<<nhb,  256, 0, stream>>>(t2, NH);

    // ---- layer 3: out2 = agg(h2@W2) + b2 ----
    k_prep_wt  <<<wb,   256, 0, stream>>>(W2, Wt);
    k_gemm     <<<gg,   256, 0, stream>>>(t2, Wt, hbuf, N);
    k_init_bias<<<nhb,  256, 0, stream>>>(out2, b2, NH);
    k_agg      <<<aggb, 256, 0, stream>>>(hbuf, row, col, dinv, out2, E, N);

    // ---- layer 4: out3 = agg(h2@W3) + b3 ----
    k_prep_wt  <<<wb,   256, 0, stream>>>(W3, Wt);
    k_gemm     <<<gg,   256, 0, stream>>>(t2, Wt, hbuf, N);
    k_init_bias<<<nhb,  256, 0, stream>>>(out3, b3, NH);
    k_agg      <<<aggb, 256, 0, stream>>>(hbuf, row, col, dinv, out3, E, N);
}
